// SelfMatcher_24146306138772
// MI455X (gfx1250) — matrix operand order repaired and checked by reading, not compiled, not run on hardware
//
#include <hip/hip_runtime.h>
#include <hip/hip_bf16.h>
#include <math.h>

// ---------------------------------------------------------------------------
// Problem constants (reference: L=512, B=64, D=H=512)
// ---------------------------------------------------------------------------
#define LL 512
#define BB 64
#define DD 512
#define HH 512
#define G3 (3 * HH)
#define KK 512   // contraction dim for every GEMM in this problem

typedef __attribute__((ext_vector_type(8)))  _Float16 v8h;
typedef __attribute__((ext_vector_type(16))) _Float16 v16h;
typedef __attribute__((ext_vector_type(8)))  float    v8f;

__device__ inline v16h pack16(v8h lo, v8h hi) {
  v16h r;
#pragma unroll
  for (int i = 0; i < 8; ++i) { r[i] = lo[i]; r[i + 8] = hi[i]; }
  return r;
}

// ---------------------------------------------------------------------------
// Tensor Data Mover staging (CDNA5).  Builds a D# descriptor (ISA 08, §8)
// for a 2D tile of 16-bit elements and issues TENSOR_LOAD_TO_LDS.
// This toolchain exposes the 6-argument builtin:
//   (u32x4 g0, i32x8 g1, i32x4 g2, i32x4 g3, i32x8 extra, i32 cpol)
// Falls back to a cooperative vector copy if the builtin is absent.
// ---------------------------------------------------------------------------
#if defined(__has_builtin)
#if __has_builtin(__builtin_amdgcn_tensor_load_to_lds) && \
    __has_builtin(__builtin_amdgcn_s_wait_tensorcnt)
#define USE_TDM 1
#endif
#endif
#ifndef USE_TDM
#define USE_TDM 0
#endif

#if USE_TDM
typedef __attribute__((ext_vector_type(4))) unsigned int u32x4;
typedef __attribute__((ext_vector_type(8))) int          i32x8;
typedef __attribute__((ext_vector_type(4))) int          i32x4;
typedef __attribute__((address_space(3)))   _Float16     lds_f16_t;

__device__ __forceinline__ void tdm_load_tile_2d_f16(
    const _Float16* gsrc,      // global tile start
    _Float16* lds_dst,         // LDS tile start
    unsigned tile_d0,          // tile width  (elements, contiguous dim)
    unsigned tile_d1,          // tile height (rows)
    unsigned tensor_d0,        // full tensor dim0 (elements)
    unsigned tensor_d1,        // full tensor dim1 (rows)
    unsigned stride0)          // row stride (elements)
{
  const unsigned long long ga = (unsigned long long)gsrc;
  const unsigned lds_off = (unsigned)(unsigned long long)(lds_f16_t*)lds_dst;

  u32x4 g0;
  g0.x = 1u;                                    // count=1, user mode, no gather
  g0.y = lds_off;                               // lds_addr (bytes)
  g0.z = (unsigned)ga;                          // global_addr[31:0]
  g0.w = (unsigned)((ga >> 32) & 0x1FFFFFFull)  // global_addr[56:32]
         | (2u << 30);                          // type = 2 ("image")

  i32x8 g1;
  g1[0] = 1 << 16;                              // data_size = 1 -> 2 bytes
  g1[1] = (int)((tensor_d0 & 0xFFFFu) << 16);   // tensor_dim0[15:0] @ bits 63:48
  g1[2] = (int)(((tensor_d0 >> 16) & 0xFFFFu) |
                ((tensor_d1 & 0xFFFFu) << 16)); // dim0 hi | tensor_dim1 lo
  g1[3] = (int)(((tensor_d1 >> 16) & 0xFFFFu) |
                ((tile_d0 & 0xFFFFu) << 16));   // dim1 hi | tile_dim0
  g1[4] = (int)(tile_d1 & 0xFFFFu);             // tile_dim1 ; tile_dim2 = 0
  g1[5] = (int)stride0;                         // tensor_dim0_stride[31:0]
  g1[6] = 0;                                    // stride0 hi | dim1_stride lo
  g1[7] = 0;

  const i32x4 z4 = {0, 0, 0, 0};                // groups 2/3: dims unused (2D)
  const i32x8 z8 = {0, 0, 0, 0, 0, 0, 0, 0};
  __builtin_amdgcn_tensor_load_to_lds(g0, g1, z4, z4, z8, 0);
}
#endif

// ---------------------------------------------------------------------------
// f32 -> f16 convert / h init
// ---------------------------------------------------------------------------
__global__ void k_cvt_f16(const float* __restrict__ src, _Float16* __restrict__ dst, int n) {
  int t = blockIdx.x * blockDim.x + threadIdx.x;
  if (t < n) dst[t] = (_Float16)src[t];
}

__global__ void k_init_h(const float* __restrict__ h0, float* __restrict__ h,
                         _Float16* __restrict__ hb, int n) {
  int t = blockIdx.x * blockDim.x + threadIdx.x;
  if (t < n) { float x = h0[t]; h[t] = x; hb[t] = (_Float16)x; }
}

// ---------------------------------------------------------------------------
// WMMA GEMM with TDM-staged B:  C[M,N] f32 = A[M,512] f16 x B[N,512]^T f16.
// Grid: gridDim.x = (N/32) * (M / (32*wavesPerBlock)); each block owns one
// 32-column strip of B for the full K=512 (32 KB LDS), staged once by the
// Tensor Data Mover, then each wave computes a 32x32 output macro-tile with
// 4 v_wmma_f32_16x16x32_f16 per 32-wide K step.  A streams from global (L2).
// M,N multiples of 32*wpb / 32 (exact for all uses).  Guards are wave-uniform
// so EXEC is all-ones for every WMMA (ISA 7.12 requirement).
// ---------------------------------------------------------------------------
__global__ __launch_bounds__(256) void k_gemm_f16(
    const _Float16* __restrict__ A, const _Float16* __restrict__ Bm,
    float* __restrict__ C, int M, int N)
{
  __shared__ _Float16 sB[32 * KK];             // 32 KB B strip: [32 rows][512 k]

  const int lane = threadIdx.x & 31;
  const int wave = threadIdx.x >> 5;
  const int wpb  = blockDim.x >> 5;
  const int n2t  = N >> 5;
  const int bn   = blockIdx.x % n2t;
  const int bm   = blockIdx.x / n2t;
  const int m2   = bm * wpb + wave;
  const int n0   = bn * 32;

  // ---- stage B strip [n0, n0+32) x [0, 512) into LDS
#if USE_TDM
  if (wave == 0) {
    tdm_load_tile_2d_f16(Bm + (size_t)n0 * KK, sB,
                         /*tile_d0=*/KK, /*tile_d1=*/32,
                         /*tensor_d0=*/KK, /*tensor_d1=*/(unsigned)N,
                         /*stride0=*/KK);
    __builtin_amdgcn_s_wait_tensorcnt(0);
  }
#else
  {
    const v8h* src = (const v8h*)(Bm + (size_t)n0 * KK);
    v8h*       dst = (v8h*)sB;
    for (int i = threadIdx.x; i < (32 * KK) / 8; i += blockDim.x) dst[i] = src[i];
  }
#endif
  __syncthreads();

  if (m2 * 32 >= M) return;                    // wave-uniform
  const int m0 = m2 * 32;

  // A fragment (16x32 f16): row = lane%16, K halves selected by lane/16
  const int rA    = lane & 15;
  const int kAsel = (lane >> 4) * 8;           // VGPR0-3 -> kAsel..+7, VGPR4-7 -> +16..+23
  // B fragment (32x16 f16): col = lane%16, same K split as A (kBsel..+7, +16..+23)
  const int cB    = lane & 15;
  const int kBsel = (lane >> 4) * 8;

  const _Float16* a0p = A + (size_t)(m0 + rA) * KK + kAsel;
  const _Float16* a1p = a0p + (size_t)16 * KK;
  const _Float16* bl0 = sB + (size_t)cB * KK + kBsel;        // LDS
  const _Float16* bl1 = bl0 + (size_t)16 * KK;               // LDS

  v8f acc00 = {}, acc01 = {}, acc10 = {}, acc11 = {};

  for (int kk = 0; kk < KK; kk += 32) {
    if (kk + 32 < KK) {                        // global_prefetch_b8 for A stream
      __builtin_prefetch(a0p + kk + 32, 0, 3);
      __builtin_prefetch(a1p + kk + 32, 0, 3);
    }
    v16h a0 = pack16(*(const v8h*)(a0p + kk), *(const v8h*)(a0p + kk + 16));
    v16h a1 = pack16(*(const v8h*)(a1p + kk), *(const v8h*)(a1p + kk + 16));
    v16h b0 = pack16(*(const v8h*)(bl0 + kk), *(const v8h*)(bl0 + kk + 16));
    v16h b1 = pack16(*(const v8h*)(bl1 + kk), *(const v8h*)(bl1 + kk + 16));

    acc00 = __builtin_amdgcn_wmma_f32_16x16x32_f16(false, a0, false, b0, (short)0, acc00, false, false);
    acc01 = __builtin_amdgcn_wmma_f32_16x16x32_f16(false, a0, false, b1, (short)0, acc01, false, false);
    acc10 = __builtin_amdgcn_wmma_f32_16x16x32_f16(false, a1, false, b0, (short)0, acc10, false, false);
    acc11 = __builtin_amdgcn_wmma_f32_16x16x32_f16(false, a1, false, b1, (short)0, acc11, false, false);
  }

  // C/D layout: lane -> col n0+(lane%16); VGPR r -> row m0 + (lane/16)*8 + r
  const int rowSel = (lane >> 4) * 8;
#pragma unroll
  for (int r = 0; r < 8; ++r) {
    const int mA = m0 + rowSel + r;
    const int mB = mA + 16;
    C[(size_t)mA * N + n0 + cB]      = acc00[r];
    C[(size_t)mA * N + n0 + 16 + cB] = acc01[r];
    C[(size_t)mB * N + n0 + cB]      = acc10[r];
    C[(size_t)mB * N + n0 + 16 + cB] = acc11[r];
  }
}

// ---------------------------------------------------------------------------
// Fused per-step attention: scores (tanh pass) -> softmax over l' -> context.
// One block per batch element b; 1024 threads = 32 waves.
// Query row Wpv[step,b,:] and V[b,:] are register-cached (16 f32/lane) and
// reused across all 512 key rows; Wp_v + vb (~100 MB) stay L2-resident.
// ---------------------------------------------------------------------------
__global__ __launch_bounds__(1024) void k_attn(
    int step,
    const float* __restrict__ Wpv,   // (L,B,H) f32
    const float* __restrict__ Wp_v,  // (L,B,H) f32
    const float* __restrict__ Vf,    // (B,H) f32
    const _Float16* __restrict__ vb, // (L,B,D) f16
    _Float16* __restrict__ cb)       // (B,D) f16 out
{
  const int b    = blockIdx.x;
  const int tid  = threadIdx.x;
  const int lane = tid & 31;
  const int wv   = tid >> 5;          // 0..31

  __shared__ float sm_s[LL];
  __shared__ float sm_red[32];

  float qv[16], vv[16];
  const float* qrow = Wpv + ((size_t)step * BB + b) * HH;
  const float* vrow = Vf + (size_t)b * HH;
#pragma unroll
  for (int k = 0; k < 16; ++k) { int hh = lane + 32 * k; qv[k] = qrow[hh]; vv[k] = vrow[hh]; }

  for (int lp = wv; lp < LL; lp += 32) {
    const float* krow = Wp_v + ((size_t)lp * BB + b) * HH;
    float s = 0.f;
#pragma unroll
    for (int k = 0; k < 16; ++k) {
      int hh = lane + 32 * k;
      s += tanhf(qv[k] + krow[hh]) * vv[k];
    }
#pragma unroll
    for (int off = 16; off > 0; off >>= 1) s += __shfl_xor(s, off, 32);
    if (lane == 0) sm_s[lp] = s;
  }
  __syncthreads();

  float m = -INFINITY;
  for (int i = tid; i < LL; i += blockDim.x) m = fmaxf(m, sm_s[i]);
#pragma unroll
  for (int off = 16; off > 0; off >>= 1) m = fmaxf(m, __shfl_xor(m, off, 32));
  if (lane == 0) sm_red[wv] = m;
  __syncthreads();
  if (wv == 0) {
    float t2 = sm_red[lane];
#pragma unroll
    for (int off = 16; off > 0; off >>= 1) t2 = fmaxf(t2, __shfl_xor(t2, off, 32));
    if (lane == 0) sm_red[0] = t2;
  }
  __syncthreads();
  m = sm_red[0];
  __syncthreads();

  float ssum = 0.f;
  for (int i = tid; i < LL; i += blockDim.x) {
    float e = __expf(sm_s[i] - m);
    sm_s[i] = e;
    ssum += e;
  }
#pragma unroll
  for (int off = 16; off > 0; off >>= 1) ssum += __shfl_xor(ssum, off, 32);
  if (lane == 0) sm_red[wv] = ssum;
  __syncthreads();
  if (wv == 0) {
    float t2 = sm_red[lane];
#pragma unroll
    for (int off = 16; off > 0; off >>= 1) t2 += __shfl_xor(t2, off, 32);
    if (lane == 0) sm_red[0] = t2;
  }
  __syncthreads();
  const float inv = 1.f / sm_red[0];
  for (int i = tid; i < LL; i += blockDim.x) sm_s[i] *= inv;
  __syncthreads();

  if (tid < DD) {
    float acc = 0.f;
    const _Float16* vcol = vb + (size_t)b * DD + tid;
#pragma unroll 4
    for (int lp = 0; lp < LL; ++lp)
      acc += sm_s[lp] * (float)vcol[(size_t)lp * BB * DD];
    cb[(size_t)b * DD + tid] = (_Float16)acc;
  }
}

// ---------------------------------------------------------------------------
// GRU gates: from gi = c@W_ih^T, gh = h@W_hh^T (no bias yet), produce h_new.
// ---------------------------------------------------------------------------
__global__ __launch_bounds__(256) void k_gates(
    const float* __restrict__ gi, const float* __restrict__ gh,
    const float* __restrict__ b_ih, const float* __restrict__ b_hh,
    float* __restrict__ h, _Float16* __restrict__ hb,
    float* __restrict__ out)
{
  const int t  = blockIdx.x * blockDim.x + threadIdx.x;  // 0..B*H-1
  const int b  = t >> 9;          // H = 512
  const int hh = t & (HH - 1);

  const float ir = gi[(size_t)b * G3 + hh]          + b_ih[hh];
  const float iz = gi[(size_t)b * G3 + HH + hh]     + b_ih[HH + hh];
  const float in = gi[(size_t)b * G3 + 2 * HH + hh] + b_ih[2 * HH + hh];
  const float hr = gh[(size_t)b * G3 + hh]          + b_hh[hh];
  const float hz = gh[(size_t)b * G3 + HH + hh]     + b_hh[HH + hh];
  const float hn = gh[(size_t)b * G3 + 2 * HH + hh] + b_hh[2 * HH + hh];

  const float r = 1.f / (1.f + __expf(-(ir + hr)));
  const float z = 1.f / (1.f + __expf(-(iz + hz)));
  const float n = tanhf(in + r * hn);
  const float hv = (1.f - z) * n + z * h[t];

  out[t] = hv;
  h[t]   = hv;
  hb[t]  = (_Float16)hv;
}

// ---------------------------------------------------------------------------
// Launch
// ---------------------------------------------------------------------------
extern "C" void kernel_launch(void* const* d_in, const int* in_sizes, int n_in,
                              void* d_out, int out_size, void* d_ws, size_t ws_size,
                              hipStream_t stream) {
  (void)in_sizes; (void)n_in; (void)out_size; (void)ws_size;
  const float* v   = (const float*)d_in[0];
  const float* h0  = (const float*)d_in[1];
  const float* Vf  = (const float*)d_in[2];
  const float* Wp  = (const float*)d_in[3];
  const float* Wp_ = (const float*)d_in[4];
  const float* Wih = (const float*)d_in[5];
  const float* Whh = (const float*)d_in[6];
  const float* bih = (const float*)d_in[7];
  const float* bhh = (const float*)d_in[8];
  float* out = (float*)d_out;

  // workspace carve-up (~173 MB total)
  char* ws = (char*)d_ws;
  auto carve = [&](size_t bytes) -> char* {
    char* p = ws; ws += (bytes + 255) & ~(size_t)255; return p;
  };
  _Float16* vb   = (_Float16*)carve((size_t)LL * BB * DD * 2);
  _Float16* Wpb  = (_Float16*)carve((size_t)HH * DD * 2);
  _Float16* Wp_b = (_Float16*)carve((size_t)HH * DD * 2);
  _Float16* Wihb = (_Float16*)carve((size_t)G3 * DD * 2);
  _Float16* Whhb = (_Float16*)carve((size_t)G3 * HH * 2);
  float*    Wpv  = (float*)   carve((size_t)LL * BB * HH * 4);
  float*    Wp_v = (float*)   carve((size_t)LL * BB * HH * 4);
  _Float16* cb   = (_Float16*)carve((size_t)BB * DD * 2);
  float*    hbuf = (float*)   carve((size_t)BB * HH * 4);
  _Float16* hb   = (_Float16*)carve((size_t)BB * HH * 2);
  float*    gi   = (float*)   carve((size_t)BB * G3 * 4);
  float*    gh   = (float*)   carve((size_t)BB * G3 * 4);

  // --- converts + h init
  {
    int n = LL * BB * DD;
    k_cvt_f16<<<(n + 255) / 256, 256, 0, stream>>>(v, vb, n);
    n = HH * DD;
    k_cvt_f16<<<(n + 255) / 256, 256, 0, stream>>>(Wp, Wpb, n);
    k_cvt_f16<<<(n + 255) / 256, 256, 0, stream>>>(Wp_, Wp_b, n);
    n = G3 * DD;
    k_cvt_f16<<<(n + 255) / 256, 256, 0, stream>>>(Wih, Wihb, n);
    n = G3 * HH;
    k_cvt_f16<<<(n + 255) / 256, 256, 0, stream>>>(Whh, Whhb, n);
    n = BB * HH;
    k_init_h<<<(n + 255) / 256, 256, 0, stream>>>(h0, hbuf, hb, n);
  }

  // --- precompute Wpv, Wp_v  (M=32768, N=512, K=512); 256 thr = 8 waves/block
  {
    const int n2t  = HH / 32;                        // 16
    const int blks = n2t * ((LL * BB) / (32 * 8));   // 16 * 128 = 2048
    k_gemm_f16<<<blks, 256, 0, stream>>>(vb, Wpb,  Wpv,  LL * BB, HH);
    k_gemm_f16<<<blks, 256, 0, stream>>>(vb, Wp_b, Wp_v, LL * BB, HH);
  }

  // --- sequential scan; GRU GEMMs use 64-thread blocks (2 waves -> M=64 exact)
  const int n2tg  = G3 / 32;                         // 48
  const int blkg  = n2tg * (BB / (32 * 2));          // 48
  for (int l = 0; l < LL; ++l) {
    k_attn<<<BB, 1024, 0, stream>>>(l, Wpv, Wp_v, Vf, vb, cb);
    k_gemm_f16<<<blkg, 64, 0, stream>>>(cb, Wihb, gi, BB, G3);
    k_gemm_f16<<<blkg, 64, 0, stream>>>(hb, Whhb, gh, BB, G3);
    k_gates<<<(BB * HH) / 256, 256, 0, stream>>>(gi, gh, bih, bhh, hbuf, hb,
                                                 out + (size_t)l * BB * HH);
  }
}
